// CNN_44504451121307
// MI455X (gfx1250) — matrix operand order repaired and checked by reading, not compiled, not run on hardware
//
#include <hip/hip_runtime.h>
#include <hip/hip_bf16.h>
#include <stdint.h>

typedef __attribute__((ext_vector_type(16))) __bf16 v16bf;
typedef __attribute__((ext_vector_type(8)))  float  v8f;
typedef unsigned int u32x4 __attribute__((ext_vector_type(4)));
typedef int          i32x8 __attribute__((ext_vector_type(8)));
typedef int          i32x4 __attribute__((ext_vector_type(4)));

#define Bn 8
#define Hh 256
#define Ww 256
#define Cm 128

// LDS staging: 4 row slots, 18 pixels each, 256B data + 16B TDM pad per pixel.
#define PXSTRIDE 272
#define SLOT     (18 * PXSTRIDE)   // 4896 B

union Frag {
  v16bf v;
  uint4 q[2];
};

__device__ __forceinline__ uint32_t bfpair(float lo, float hi) {
  uint32_t ua = __float_as_uint(lo);
  uint32_t ub = __float_as_uint(hi);
  uint32_t l = (ua + 0x7FFFu + ((ua >> 16) & 1u)) >> 16;
  uint32_t h = (ub + 0x7FFFu + ((ub >> 16) & 1u)) & 0xFFFF0000u;
  return l | h;
}

// ---------------------------------------------------------------------------
// TDM: async-load one contiguous strip (n_elem4 dwords) from global into LDS,
// inserting 4 dwords of pad after every 64 dwords (=> 272B pixel stride).
// D# per CDNA5 ISA.
// ---------------------------------------------------------------------------
__device__ __forceinline__ void tdm_load_strip(const void* gsrc, uint32_t lds_byte,
                                               uint32_t n_elem4) {
  uint64_t ga = (uint64_t)(uintptr_t)gsrc;
  u32x4 g0;
  g0[0] = 1u;                                   // count=1, load, user mode
  g0[1] = lds_byte;                             // lds_addr
  g0[2] = (uint32_t)ga;                         // global_addr[31:0]
  g0[3] = (uint32_t)((ga >> 32) & 0x01FFFFFFu)  // global_addr[56:32]
          | 0x80000000u;                        // type=2 ("image")
  i32x8 g1;
  g1[0] = (int)((2u << 16)      // data_size = 4B
              | (1u << 20)      // pad_enable
              | (5u << 22)      // pad_interval: 64 dwords (256B)
              | (3u << 25));    // pad_amount: 4 dwords (16B)
  g1[1] = (int)((n_elem4 & 0xFFFFu) << 16);                  // tensor_dim0 lo
  g1[2] = (int)(((n_elem4 >> 16) & 0xFFFFu) | (1u << 16));   // dim0 hi | tensor_dim1=1
  g1[3] = (int)(n_elem4 << 16);                              // tile_dim0
  g1[4] = 0;                                                 // tile_dim1/2 unused
  g1[5] = (int)n_elem4;                                      // tensor_dim0_stride
  g1[6] = 0;
  g1[7] = 0;
  i32x4 z4 = {0, 0, 0, 0};
#if __clang_major__ >= 23
  i32x8 z8 = {0, 0, 0, 0, 0, 0, 0, 0};
  __builtin_amdgcn_tensor_load_to_lds(g0, g1, z4, z4, z8, 0);
#else
  __builtin_amdgcn_tensor_load_to_lds(g0, g1, z4, z4, 0);
#endif
}

// ---------------------------------------------------------------------------
// Pre-pack 128x128x5 f32 weights into A fragments (contiguous k = 16*half + s, same order as B) of
// V_WMMA_F32_16X16X32_BF16. Buffer layout: [tap][kb][octile][lane][8 dwords].
// ---------------------------------------------------------------------------
__global__ void pack_w_mid(const float* __restrict__ w, uint32_t* __restrict__ apack) {
  int tid = blockIdx.x * blockDim.x + threadIdx.x;
  if (tid >= 5 * 4 * 8 * 32) return;
  int lane   = tid & 31;
  int octile = (tid >> 5) & 7;
  int kb     = (tid >> 8) & 3;
  int tap    = tid >> 10;
  int l2 = lane & 15, half = lane >> 4;
  int oc = octile * 16 + l2;
  uint32_t* dst = apack + (size_t)tid * 8;
#pragma unroll
  for (int v = 0; v < 8; ++v) {
    int kl = half * 16 + 2 * v;
    int k0 = kb * 32 + kl;
    float f0 = w[((size_t)oc * Cm + k0) * 5 + tap];
    float f1 = w[((size_t)oc * Cm + k0 + 1) * 5 + tap];
    dst[v] = bfpair(f0, f1);
  }
}

// ---------------------------------------------------------------------------
// Mid layer (128 -> 128, cross stencil, bias + ReLU) as implicit GEMM.
// Block = (batch, 16-px x-group, 32-row y-chunk); 8 waves = 8 oc-tiles.
// Activations staged once per block into LDS by the Tensor Data Mover with a
// 4-slot rolling row window (rows y-1..y+1 live, y+2 in flight). Halo pixel
// columns are pre-zeroed so left/right taps are unconditional ds_load_b128;
// up/down taps use a uniform scalar skip (load+wmma) at image borders.
// Two accumulator chains hide the bf16 WMMA->WMMA RAW hazard (ISA 7.12.1).
// ---------------------------------------------------------------------------
__global__ __launch_bounds__(256) void mid_layer(
    const uint16_t* __restrict__ Hin, uint16_t* __restrict__ Hout,
    const uint32_t* __restrict__ apack, const float* __restrict__ bias)
{
  __shared__ __align__(16) uint8_t smem[4 * SLOT];

  const int lane   = threadIdx.x & 31;
  const int octile = threadIdx.x >> 5;
  const int l2 = lane & 15, half = lane >> 4;

  const int blk = blockIdx.x;
  const int xg = blk & 15;
  const int yc = (blk >> 4) & 7;
  const int b  = blk >> 7;

  const int x0 = xg * 16;

  // Strip geometry for this block (uniform): pixels x0-1 .. x0+16, clamped.
  const int xs  = (x0 - 1 < 0) ? 0 : x0 - 1;
  const int xe  = (x0 + 16 > Ww - 1) ? Ww - 1 : x0 + 16;
  const uint32_t n_elem4 = (uint32_t)(xe - xs + 1) * 64;   // dwords per strip
  const uint32_t lds_px0 = (uint32_t)(xs - (x0 - 1));      // slot-local pixel index

  const uint32_t smbase = (uint32_t)(uintptr_t)(void*)smem;

  // Pre-zero the halo pixel columns (xl=0 and xl=17) of all 4 slots. TDM only
  // writes them when they are real image pixels, so out-of-image taps read 0.
  {
    const uint4 z = make_uint4(0, 0, 0, 0);
    for (int i = threadIdx.x; i < 4 * 2 * 17; i += 256) {
      int slot = i / 34;
      int rem  = i - slot * 34;
      int col  = (rem >= 17) ? 17 : 0;
      int q    = (rem >= 17) ? rem - 17 : rem;
      ((uint4*)(smem + slot * SLOT + col * PXSTRIDE))[q] = z;
    }
  }
  __syncthreads();

  // Load all 20 prepacked A fragments into registers (stay resident).
  Frag A[5][4];
#pragma unroll
  for (int t = 0; t < 5; ++t)
#pragma unroll
    for (int kb = 0; kb < 4; ++kb) {
      const uint4* s =
          (const uint4*)(apack + ((((size_t)t * 4 + kb) * 8 + octile) * 32 + lane) * 8);
      A[t][kb].q[0] = s[0];
      A[t][kb].q[1] = s[1];
    }

  float bv[8];
#pragma unroll
  for (int r = 0; r < 8; ++r) bv[r] = bias[octile * 16 + half * 8 + r];

  // Prologue: TDM rows y0-1, y0, y0+1 into their slots, wait, barrier.
  const int y0 = yc * 32;
  if (threadIdx.x < 32) {
    for (int r = y0 - 1; r <= y0 + 1; ++r) {
      if (r >= 0 && r < Hh) {
        tdm_load_strip(Hin + ((size_t)(b * Hh + r) * Ww + xs) * Cm,
                       smbase + (uint32_t)(r & 3) * SLOT + lds_px0 * PXSTRIDE,
                       n_elem4);
      }
    }
    __builtin_amdgcn_s_wait_tensorcnt(0);
  }
  __syncthreads();

  for (int yy = 0; yy < 32; ++yy) {
    const int y = y0 + yy;

    // Kick off the async TDM load of row y+2 (overwrites the y-2 slot).
    if (threadIdx.x < 32) {
      const int r = y + 2;
      if (r < Hh && r <= y0 + 32) {
        tdm_load_strip(Hin + ((size_t)(b * Hh + r) * Ww + xs) * Cm,
                       smbase + (uint32_t)(r & 3) * SLOT + lds_px0 * PXSTRIDE,
                       n_elem4);
      }
    }

    v8f acc0, acc1;
#pragma unroll
    for (int r = 0; r < 8; ++r) { acc0[r] = bv[r]; acc1[r] = 0.f; }

    const uint32_t baseC =
        (uint32_t)(y & 3) * SLOT + (uint32_t)(l2 + 1) * PXSTRIDE + half * 32;
    const uint32_t baseU =
        (uint32_t)((y - 1) & 3) * SLOT + (uint32_t)(l2 + 1) * PXSTRIDE + half * 32;
    const uint32_t baseD =
        (uint32_t)((y + 1) & 3) * SLOT + (uint32_t)(l2 + 1) * PXSTRIDE + half * 32;
    const bool upok = (y > 0);
    const bool dnok = (y < Hh - 1);

#pragma unroll
    for (int kb = 0; kb < 4; ++kb) {
      Frag Bf;
      // tap 0: center  -> chain 0
      Bf.q[0] = ((const uint4*)(smem + baseC + kb * 64))[0];
      Bf.q[1] = ((const uint4*)(smem + baseC + kb * 64))[1];
      acc0 = __builtin_amdgcn_wmma_f32_16x16x32_bf16(
          false, A[0][kb].v, false, Bf.v, (short)0, acc0, false, false);
      // tap 3: left    -> chain 1 (halo column pre-zeroed at image edge)
      Bf.q[0] = ((const uint4*)(smem + baseC - PXSTRIDE + kb * 64))[0];
      Bf.q[1] = ((const uint4*)(smem + baseC - PXSTRIDE + kb * 64))[1];
      acc1 = __builtin_amdgcn_wmma_f32_16x16x32_bf16(
          false, A[3][kb].v, false, Bf.v, (short)0, acc1, false, false);
      // tap 4: right   -> chain 0
      Bf.q[0] = ((const uint4*)(smem + baseC + PXSTRIDE + kb * 64))[0];
      Bf.q[1] = ((const uint4*)(smem + baseC + PXSTRIDE + kb * 64))[1];
      acc0 = __builtin_amdgcn_wmma_f32_16x16x32_bf16(
          false, A[4][kb].v, false, Bf.v, (short)0, acc0, false, false);
      // tap 1: up      -> chain 1, uniform skip at top border
      if (upok) {
        Bf.q[0] = ((const uint4*)(smem + baseU + kb * 64))[0];
        Bf.q[1] = ((const uint4*)(smem + baseU + kb * 64))[1];
        acc1 = __builtin_amdgcn_wmma_f32_16x16x32_bf16(
            false, A[1][kb].v, false, Bf.v, (short)0, acc1, false, false);
      }
      // tap 2: down    -> chain 0, uniform skip at bottom border
      if (dnok) {
        Bf.q[0] = ((const uint4*)(smem + baseD + kb * 64))[0];
        Bf.q[1] = ((const uint4*)(smem + baseD + kb * 64))[1];
        acc0 = __builtin_amdgcn_wmma_f32_16x16x32_bf16(
            false, A[2][kb].v, false, Bf.v, (short)0, acc0, false, false);
      }
    }

    // C/D layout: VGPR r, lane l -> oc = octile*16 + 8*half + r, pixel = l2.
    uint4 o;
    o.x = bfpair(fmaxf(acc0[0] + acc1[0], 0.f), fmaxf(acc0[1] + acc1[1], 0.f));
    o.y = bfpair(fmaxf(acc0[2] + acc1[2], 0.f), fmaxf(acc0[3] + acc1[3], 0.f));
    o.z = bfpair(fmaxf(acc0[4] + acc1[4], 0.f), fmaxf(acc0[5] + acc1[5], 0.f));
    o.w = bfpair(fmaxf(acc0[6] + acc1[6], 0.f), fmaxf(acc0[7] + acc1[7], 0.f));
    uint16_t* dst =
        Hout + ((size_t)(b * Hh + y) * Ww + (x0 + l2)) * Cm + octile * 16 + half * 8;
    *(uint4*)dst = o;

    if (threadIdx.x < 32) __builtin_amdgcn_s_wait_tensorcnt(0);
    __syncthreads();
  }
}

// ---------------------------------------------------------------------------
// Layer 1: 6 -> 128, f32 VALU (2% of FLOPs), writes bf16 channel-last.
// ---------------------------------------------------------------------------
__global__ __launch_bounds__(256) void layer1(
    const float* __restrict__ x, const float* __restrict__ w1,
    const float* __restrict__ b1, uint16_t* __restrict__ Hout)
{
  __shared__ float wl[128 * 6 * 5];
  __shared__ float bl[128];
  for (int i = threadIdx.x; i < 128 * 6 * 5; i += 256) wl[i] = w1[i];
  for (int i = threadIdx.x; i < 128; i += 256) bl[i] = b1[i];
  __syncthreads();

  const int b  = blockIdx.x >> 8;
  const int y  = blockIdx.x & 255;
  const int xx = threadIdx.x;

  const int dy[5] = {0, -1, 1, 0, 0};
  const int dx[5] = {0, 0, 0, -1, 1};

  float xin[6][5];
#pragma unroll
  for (int ic = 0; ic < 6; ++ic)
#pragma unroll
    for (int t = 0; t < 5; ++t) {
      int py = y + dy[t], pxs = xx + dx[t];
      bool valid = (py >= 0) & (py < Hh) & (pxs >= 0) & (pxs < Ww);
      xin[ic][t] = valid ? x[(((size_t)b * 6 + ic) * Hh + py) * Ww + pxs] : 0.f;
    }

  uint32_t* outp = (uint32_t*)(Hout + ((size_t)(b * Hh + y) * Ww + xx) * Cm);
  for (int oc = 0; oc < 128; oc += 2) {
    float s0 = bl[oc], s1 = bl[oc + 1];
#pragma unroll
    for (int ic = 0; ic < 6; ++ic)
#pragma unroll
      for (int t = 0; t < 5; ++t) {
        s0 = fmaf(wl[(oc * 6 + ic) * 5 + t], xin[ic][t], s0);
        s1 = fmaf(wl[((oc + 1) * 6 + ic) * 5 + t], xin[ic][t], s1);
      }
    outp[oc >> 1] = bfpair(fmaxf(s0, 0.f), fmaxf(s1, 0.f));
  }
}

// ---------------------------------------------------------------------------
// Layer 4: 128 -> 6, no ReLU, f32 output NCHW.
// ---------------------------------------------------------------------------
__global__ __launch_bounds__(256) void layer4(
    const uint16_t* __restrict__ Hin, const float* __restrict__ w4,
    const float* __restrict__ b4, float* __restrict__ out)
{
  __shared__ float wl[5 * 128 * 6];
  __shared__ float bl[6];
  for (int i = threadIdx.x; i < 6 * 128 * 5; i += 256) {
    int oc = i / 640;
    int ic = (i / 5) % 128;
    int t  = i % 5;
    wl[(t * 128 + ic) * 6 + oc] = w4[i];
  }
  if (threadIdx.x < 6) bl[threadIdx.x] = b4[threadIdx.x];
  __syncthreads();

  const int b  = blockIdx.x >> 8;
  const int y  = blockIdx.x & 255;
  const int xx = threadIdx.x;

  const int dy[5] = {0, -1, 1, 0, 0};
  const int dx[5] = {0, 0, 0, -1, 1};

  float acc[6];
#pragma unroll
  for (int oc = 0; oc < 6; ++oc) acc[oc] = bl[oc];

#pragma unroll
  for (int t = 0; t < 5; ++t) {
    int py = y + dy[t], pxs = xx + dx[t];
    bool valid = (py >= 0) & (py < Hh) & (pxs >= 0) & (pxs < Ww);
    if (valid) {
      const uint4* src =
          (const uint4*)(Hin + ((size_t)(b * Hh + py) * Ww + pxs) * Cm);
#pragma unroll 4
      for (int c16 = 0; c16 < 16; ++c16) {
        uint4 q = src[c16];
        uint32_t u[4] = {q.x, q.y, q.z, q.w};
#pragma unroll
        for (int j = 0; j < 4; ++j) {
          float f0 = __uint_as_float(u[j] << 16);
          float f1 = __uint_as_float(u[j] & 0xFFFF0000u);
          int ic = c16 * 8 + j * 2;
          const float* wp0 = &wl[(t * 128 + ic) * 6];
          const float* wp1 = wp0 + 6;
#pragma unroll
          for (int oc = 0; oc < 6; ++oc) {
            acc[oc] = fmaf(wp0[oc], f0, acc[oc]);
            acc[oc] = fmaf(wp1[oc], f1, acc[oc]);
          }
        }
      }
    }
  }
#pragma unroll
  for (int oc = 0; oc < 6; ++oc)
    out[(((size_t)b * 6 + oc) * Hh + y) * Ww + xx] = acc[oc];
}

// ---------------------------------------------------------------------------
extern "C" void kernel_launch(void* const* d_in, const int* in_sizes, int n_in,
                              void* d_out, int out_size, void* d_ws, size_t ws_size,
                              hipStream_t stream)
{
  const float* x  = (const float*)d_in[0];
  const float* w1 = (const float*)d_in[1];
  const float* b1 = (const float*)d_in[2];
  const float* w2 = (const float*)d_in[3];
  const float* b2 = (const float*)d_in[4];
  const float* w3 = (const float*)d_in[5];
  const float* b3 = (const float*)d_in[6];
  const float* w4 = (const float*)d_in[7];
  const float* b4 = (const float*)d_in[8];
  float* out = (float*)d_out;

  const size_t hbytes = (size_t)Bn * Hh * Ww * Cm * sizeof(uint16_t);
  uint16_t* hA  = (uint16_t*)d_ws;
  uint16_t* hB  = (uint16_t*)((char*)d_ws + hbytes);
  uint32_t* ap2 = (uint32_t*)((char*)d_ws + 2 * hbytes);
  uint32_t* ap3 = ap2 + (size_t)5 * 4 * 8 * 32 * 8;

  pack_w_mid<<<20, 256, 0, stream>>>(w2, ap2);
  pack_w_mid<<<20, 256, 0, stream>>>(w3, ap3);
  layer1<<<Bn * Hh, 256, 0, stream>>>(x, w1, b1, hA);
  mid_layer<<<1024, 256, 0, stream>>>(hA, hB, ap2, b2);
  mid_layer<<<1024, 256, 0, stream>>>(hB, hA, ap3, b3);
  layer4<<<Bn * Hh, 256, 0, stream>>>(hA, w4, b4, out);
}
